// CriticNet_59399397704010
// MI455X (gfx1250) — matrix operand order repaired and checked by reading, not compiled, not run on hardware
//
#include <hip/hip_runtime.h>
#include <hip/hip_bf16.h>

// Sizes from the reference
#define B_N 32768
#define G_N 2048
#define H_N 512
#define A_N 8
#define PROJ_HID 64
#define HEAD_HID 32

typedef __attribute__((ext_vector_type(16))) __bf16 v16bf;
typedef __attribute__((ext_vector_type(8)))  float  v8f;

union Frag {
    v16bf v;
    unsigned int u[8];
    unsigned short us[16];
};

__device__ __forceinline__ unsigned short f2bf(float f) {
    unsigned int u = __float_as_uint(f);
    u += 0x7FFFu + ((u >> 16) & 1u);   // round-to-nearest-even
    return (unsigned short)(u >> 16);
}

// 16-bit A-matrix 16x32 K index for VGPR-dword j (2 bf16 per dword), lane half.
// CDNA5 ISA: VGPR0..3 -> K={0..7|8..15}, VGPR4..7 -> K={16..23|24..31}
__device__ __forceinline__ int a_kof(int j, int half) {
    return (j < 4) ? (half * 8 + 2 * j) : (16 + half * 8 + 2 * (j - 4));
}

__device__ __forceinline__ v8f wmma_bf16(const Frag& a, const Frag& b, v8f c) {
    return __builtin_amdgcn_wmma_f32_16x16x32_bf16(
        false, a.v, false, b.v, (short)0, c, false, false);
}

__device__ __forceinline__ float sigmoidf_(float x) {
    return 1.0f / (1.0f + __expf(-x));
}

// ---------------- conversion kernels -------------------------------------------------

// f32 -> bf16, same layout
__global__ void k_cvt(const float* __restrict__ src, unsigned short* __restrict__ dst, int n) {
    int i = blockIdx.x * blockDim.x + threadIdx.x;
    if (i < n) dst[i] = f2bf(src[i]);
}

// f32 [R][C] -> bf16 [C][R] (K-major B operand for "x @ W^T" GEMMs)
__global__ void k_cvt_T(const float* __restrict__ src, unsigned short* __restrict__ dst,
                        int R, int C) {
    int i = blockIdx.x * blockDim.x + threadIdx.x;
    if (i < R * C) {
        int r = i / C, c = i % C;
        dst[c * R + r] = f2bf(src[i]);
    }
}

// head_w1 [A][H][32] f32 -> bf16 [H][A*32] (K-major, agents concatenated on N)
__global__ void k_cvt_headw1(const float* __restrict__ src, unsigned short* __restrict__ dst) {
    int i = blockIdx.x * blockDim.x + threadIdx.x;
    if (i < A_N * H_N * HEAD_HID) {
        int a = i / (H_N * HEAD_HID);
        int rem = i - a * (H_N * HEAD_HID);
        int k = rem / HEAD_HID;
        int c = rem % HEAD_HID;
        dst[k * (A_N * HEAD_HID) + a * HEAD_HID + c] = f2bf(src[i]);
    }
}

// ---------------- proj stage 1: hidden = relu(obs @ w0 + b0) : [B,64] bf16 -----------
// grid.x = B/128, 256 threads (8 waves); wave w -> rows 16w..16w+15, full N=64 (4 tiles)
// Streams the 268MB obs matrix: b128 global loads, f32->bf16 convert into LDS.
__global__ void k_proj1(const float* __restrict__ obs,
                        const unsigned short* __restrict__ w0bf,  // [G][64] k-major bf16
                        const float* __restrict__ b0,
                        unsigned short* __restrict__ hid_bf) {    // [B][64] bf16
    __shared__ unsigned short As[128 * 32];   // 8 KB
    __shared__ unsigned short Bs[32 * 64];    // 4 KB

    const int t = threadIdx.x;
    const int w = t >> 5, l = t & 31;
    const int half = l >> 4, m = l & 15;
    const int row0 = blockIdx.x * 128;

    v8f acc[4];
#pragma unroll
    for (int i = 0; i < 4; ++i) acc[i] = (v8f){};

    for (int k0 = 0; k0 < G_N; k0 += 32) {
        // A tile: 128x32 f32 -> bf16 into LDS, 16B global loads
        for (int i = t; i < 128 * 8; i += 256) {          // 1024 float4's
            int r = i >> 3;
            int c4 = (i & 7) << 2;                         // 0,4,...,28
            float4 f = *(const float4*)(obs + (size_t)(row0 + r) * G_N + k0 + c4);
            unsigned int lo = (unsigned int)f2bf(f.x) | ((unsigned int)f2bf(f.y) << 16);
            unsigned int hi = (unsigned int)f2bf(f.z) | ((unsigned int)f2bf(f.w) << 16);
            unsigned int* dst = (unsigned int*)&As[r * 32 + c4];
            dst[0] = lo;
            dst[1] = hi;
        }
        // B tile: rows k0..k0+31 of w0bf (contiguous 32*64 bf16), copy as dwords
        {
            const unsigned int* src = (const unsigned int*)(w0bf + (size_t)k0 * 64);
            unsigned int* dst = (unsigned int*)Bs;
            for (int i = t; i < 32 * 64 / 2; i += 256) dst[i] = src[i];
        }
        __syncthreads();

        Frag a;
#pragma unroll
        for (int j = 0; j < 8; ++j)
            a.u[j] = *(const unsigned int*)&As[(16 * w + m) * 32 + a_kof(j, half)];

#pragma unroll
        for (int tl = 0; tl < 4; ++tl) {
            Frag b;  // lane = N col (l&15), dword j holds K = a_kof(j,half), +1
#pragma unroll
            for (int j = 0; j < 8; ++j) {
                int kk = a_kof(j, half);
                b.us[2 * j]     = Bs[kk * 64 + 16 * tl + m];
                b.us[2 * j + 1] = Bs[(kk + 1) * 64 + 16 * tl + m];
            }
            acc[tl] = wmma_bf16(a, b, acc[tl]);
        }
        __syncthreads();
    }

    // epilogue: +bias, relu, bf16 store. C layout: vgpr i -> M=i+8*half, N=l&15
    const int n_l = l & 15;
#pragma unroll
    for (int tl = 0; tl < 4; ++tl) {
        int col = 16 * tl + n_l;
        float bias = b0[col];
#pragma unroll
        for (int i = 0; i < 8; ++i) {
            int grow = row0 + 16 * w + i + 8 * half;
            float v = acc[tl][i] + bias;
            v = v > 0.0f ? v : 0.0f;
            hid_bf[(size_t)grow * 64 + col] = f2bf(v);
        }
    }
}

// ---------------- proj stage 2: x = hidden @ w1 + b1 : [B,512] bf16 ------------------
// grid = (B/128, 512/64), 256 threads; wave w -> 16 rows x 64 cols (4 tiles), K=64
__global__ void k_proj2(const unsigned short* __restrict__ hid_bf,  // [B][64]
                        const unsigned short* __restrict__ w1bf,    // [64][512] k-major
                        const float* __restrict__ b1,
                        unsigned short* __restrict__ x_bf) {        // [B][512]
    const int t = threadIdx.x;
    const int w = t >> 5, l = t & 31;
    const int half = l >> 4, m = l & 15;
    const int row0 = blockIdx.x * 128;
    const int col0 = blockIdx.y * 64;
    const int grow_a = row0 + 16 * w + m;

    v8f acc[4];
#pragma unroll
    for (int i = 0; i < 4; ++i) acc[i] = (v8f){};

#pragma unroll
    for (int k0 = 0; k0 < 64; k0 += 32) {
        Frag a;
#pragma unroll
        for (int j = 0; j < 8; ++j)
            a.u[j] = *(const unsigned int*)&hid_bf[(size_t)grow_a * 64 + k0 + a_kof(j, half)];
#pragma unroll
        for (int tl = 0; tl < 4; ++tl) {
            Frag b;
#pragma unroll
            for (int j = 0; j < 8; ++j) {
                int kk = k0 + a_kof(j, half);
                b.us[2 * j]     = w1bf[(size_t)kk * H_N + col0 + 16 * tl + m];
                b.us[2 * j + 1] = w1bf[(size_t)(kk + 1) * H_N + col0 + 16 * tl + m];
            }
            acc[tl] = wmma_bf16(a, b, acc[tl]);
        }
    }

    const int n_l = l & 15;
#pragma unroll
    for (int tl = 0; tl < 4; ++tl) {
        int col = col0 + 16 * tl + n_l;
        float bias = b1[col];
#pragma unroll
        for (int i = 0; i < 8; ++i) {
            int grow = row0 + 16 * w + i + 8 * half;
            x_bf[(size_t)grow * H_N + col] = f2bf(acc[tl][i] + bias);
        }
    }
}

// ---------------- fused GRU step -----------------------------------------------------
// gi = x@Wih^T + b_ih ; gh = h_prev@Whh^T + b_hh ; gate math fused in registers.
// grid = (B/128, 512/32), 256 threads; wave w -> rows 16w, block cols = 32 (2 tiles).
// 12 WMMA accumulator tiles per wave: {i,h} x {r,z,n} x 2 col tiles. gi/gh never in memory.
// Gate weight sub-tiles staged once per block in LDS (shared by all 8 waves).
__global__ void k_gru(const unsigned short* __restrict__ x_bf,   // [B][512] bf16
                      const unsigned short* __restrict__ hp_bf,  // [B][512] bf16
                      const float* __restrict__ h_prev,          // [B][512] f32 (epilogue)
                      const unsigned short* __restrict__ wihT,   // [512][1536] k-major
                      const unsigned short* __restrict__ whhT,   // [512][1536] k-major
                      const float* __restrict__ b_ih,            // [1536]
                      const float* __restrict__ b_hh,            // [1536]
                      float* __restrict__ h_out,                 // [B][512] -> d_out+B
                      unsigned short* __restrict__ h_bf) {       // [B][512] bf16 for heads
    // [mat 2][gate 3][k 32][nd 16] dwords = 2 bf16 per dword, n = 2*nd
    __shared__ unsigned short Bs[2 * 3 * 32 * 32];   // 12 KB

    const int t = threadIdx.x;
    const int w = t >> 5, l = t & 31;
    const int half = l >> 4, m = l & 15;
    const int r0 = blockIdx.x * 128 + 16 * w;
    const int c0 = blockIdx.y * 32;
    const size_t arow = (size_t)(r0 + m) * H_N;

    v8f accI[2][3], accH[2][3];
#pragma unroll
    for (int ct = 0; ct < 2; ++ct)
#pragma unroll
        for (int g = 0; g < 3; ++g) { accI[ct][g] = (v8f){}; accH[ct][g] = (v8f){}; }

    unsigned int* Bs_dw = (unsigned int*)Bs;

    for (int k0 = 0; k0 < H_N; k0 += 32) {
        // stage both weight matrices' 3 gate sub-tiles: 2 x 1536 dwords
        for (int i = t; i < 1536; i += 256) {
            int g = i >> 9;               // /512
            int r = i & 511;
            int k = r >> 4;
            int nd = r & 15;
            size_t src = (size_t)(k0 + k) * (3 * H_N) + g * H_N + c0 + 2 * nd;
            Bs_dw[i]        = *(const unsigned int*)&wihT[src];
            Bs_dw[1536 + i] = *(const unsigned int*)&whhT[src];
        }
        __syncthreads();

        Frag ax, ah;
#pragma unroll
        for (int j = 0; j < 8; ++j) {
            int k = a_kof(j, half);
            ax.u[j] = *(const unsigned int*)&x_bf[arow + k0 + k];
            ah.u[j] = *(const unsigned int*)&hp_bf[arow + k0 + k];
        }

#pragma unroll
        for (int g = 0; g < 3; ++g) {
#pragma unroll
            for (int ct = 0; ct < 2; ++ct) {
                Frag bi, bh;
#pragma unroll
                for (int j = 0; j < 8; ++j) {
                    int idx = g * 1024 + a_kof(j, half) * 32 + ct * 16 + m;   // [g][k][n] ushort
                    bi.us[2 * j]     = Bs[idx];
                    bi.us[2 * j + 1] = Bs[idx + 32];
                    bh.us[2 * j]     = Bs[3072 + idx];
                    bh.us[2 * j + 1] = Bs[3072 + idx + 32];
                }
                accI[ct][g] = wmma_bf16(ax, bi, accI[ct][g]);
                accH[ct][g] = wmma_bf16(ah, bh, accH[ct][g]);
            }
        }
        __syncthreads();
    }

    // gate epilogue, per-element (C layout: vgpr i -> M=i+8*half, N=l&15)
    const int n_l = l & 15;
#pragma unroll
    for (int ct = 0; ct < 2; ++ct) {
        const int col = c0 + ct * 16 + n_l;
        const float bir = b_ih[col],           bhr = b_hh[col];
        const float biz = b_ih[H_N + col],     bhz = b_hh[H_N + col];
        const float bin = b_ih[2 * H_N + col], bhn = b_hh[2 * H_N + col];
#pragma unroll
        for (int i = 0; i < 8; ++i) {
            int grow = r0 + i + 8 * half;
            float r = sigmoidf_(accI[ct][0][i] + bir + accH[ct][0][i] + bhr);
            float z = sigmoidf_(accI[ct][1][i] + biz + accH[ct][1][i] + bhz);
            float n = tanhf(accI[ct][2][i] + bin + r * (accH[ct][2][i] + bhn));
            float hp = h_prev[(size_t)grow * H_N + col];
            float h = (1.0f - z) * n + z * hp;
            h_out[(size_t)grow * H_N + col] = h;
            h_bf[(size_t)grow * H_N + col] = f2bf(h);
        }
    }
}

// ---------------- per-agent heads ----------------------------------------------------
// hid = relu(h @ headW1 + b1) for all 8 agents -> [64 rows][256] tile in LDS,
// then out[row] = dot(hid[row, a*32:(a+1)*32], w2[a]) + b2[a], a = agent_ids[row].
// grid = B/64, 256 threads = 8 waves as 4(M) x 2(N). Weight tile staged in LDS.
__global__ void k_heads(const unsigned short* __restrict__ h_bf,  // [B][512]
                        const unsigned short* __restrict__ w1h,   // [512][256] k-major
                        const float* __restrict__ b1h,            // [A][32]
                        const float* __restrict__ w2h,            // [A][32]
                        const float* __restrict__ b2h,            // [A]
                        const int* __restrict__ agent_ids,
                        float* __restrict__ out) {                // [B]
    __shared__ float hidS[64 * 256];                 // 64 KB
    __shared__ unsigned short BsH[32 * 256];         // 16 KB, [k 32][nd 128] dwords

    const int t = threadIdx.x;
    const int w = t >> 5, l = t & 31;
    const int wy = w >> 1, wx = w & 1;
    const int half = l >> 4, m = l & 15;
    const int row0 = blockIdx.x * 64;
    const size_t arow = (size_t)(row0 + 16 * wy + m) * H_N;
    const int colbase = 128 * wx;
    unsigned int* Bs_dw = (unsigned int*)BsH;

    v8f acc[8];
#pragma unroll
    for (int i = 0; i < 8; ++i) acc[i] = (v8f){};

    for (int k0 = 0; k0 < H_N; k0 += 32) {
        // stage B tile [32 k][256 n] bf16 = 4096 dwords
        for (int i = t; i < 4096; i += 256) {
            int k = i >> 7;
            int nd = i & 127;
            Bs_dw[i] = *(const unsigned int*)&w1h[(size_t)(k0 + k) * 256 + 2 * nd];
        }
        __syncthreads();

        Frag a;
#pragma unroll
        for (int j = 0; j < 8; ++j)
            a.u[j] = *(const unsigned int*)&h_bf[arow + k0 + a_kof(j, half)];

#pragma unroll
        for (int tl = 0; tl < 8; ++tl) {
            Frag b;
#pragma unroll
            for (int j = 0; j < 8; ++j) {
                int kk = a_kof(j, half);
                b.us[2 * j]     = BsH[kk * 256 + colbase + 16 * tl + m];
                b.us[2 * j + 1] = BsH[(kk + 1) * 256 + colbase + 16 * tl + m];
            }
            acc[tl] = wmma_bf16(a, b, acc[tl]);
        }
        __syncthreads();
    }

    const int n_l = l & 15;
#pragma unroll
    for (int tl = 0; tl < 8; ++tl) {
        int col = colbase + 16 * tl + n_l;        // 0..255 = a*32 + c
        float bias = b1h[col];                    // [A][32] flat
#pragma unroll
        for (int i = 0; i < 8; ++i) {
            int lrow = 16 * wy + i + 8 * half;
            float v = acc[tl][i] + bias;
            hidS[lrow * 256 + col] = v > 0.0f ? v : 0.0f;
        }
    }
    __syncthreads();

    if (t < 64) {
        int grow = row0 + t;
        int a = agent_ids[grow];
        const float* hv = &hidS[t * 256 + a * HEAD_HID];
        const float* wv = &w2h[a * HEAD_HID];
        float s = b2h[a];
#pragma unroll
        for (int c = 0; c < HEAD_HID; ++c) s += hv[c] * wv[c];
        out[grow] = s;
    }
}

// ---------------- host launcher ------------------------------------------------------

extern "C" void kernel_launch(void* const* d_in, const int* in_sizes, int n_in,
                              void* d_out, int out_size, void* d_ws, size_t ws_size,
                              hipStream_t stream) {
    const float* global_obs = (const float*)d_in[0];
    const float* h_prev     = (const float*)d_in[1];
    const int*   agent_ids  = (const int*)d_in[2];
    const float* proj_w0    = (const float*)d_in[3];
    const float* proj_b0    = (const float*)d_in[4];
    const float* proj_w1    = (const float*)d_in[5];
    const float* proj_b1    = (const float*)d_in[6];
    const float* gru_w_ih   = (const float*)d_in[7];
    const float* gru_w_hh   = (const float*)d_in[8];
    const float* gru_b_ih   = (const float*)d_in[9];
    const float* gru_b_hh   = (const float*)d_in[10];
    const float* head_w1    = (const float*)d_in[11];
    const float* head_b1    = (const float*)d_in[12];
    const float* head_w2    = (const float*)d_in[13];
    const float* head_b2    = (const float*)d_in[14];

    float* out_sel = (float*)d_out;            // [B]
    float* h_out   = (float*)d_out + B_N;      // [B][512]

    // workspace carve-up (256B aligned)
    char* ws = (char*)d_ws;
    size_t off = 0;
    auto carve = [&](size_t bytes) {
        char* p = ws + off;
        off += (bytes + 255) & ~(size_t)255;
        return p;
    };
    unsigned short* w0bf  = (unsigned short*)carve((size_t)G_N * PROJ_HID * 2);
    unsigned short* w1bf  = (unsigned short*)carve((size_t)PROJ_HID * H_N * 2);
    unsigned short* wihT  = (unsigned short*)carve((size_t)H_N * 3 * H_N * 2);
    unsigned short* whhT  = (unsigned short*)carve((size_t)H_N * 3 * H_N * 2);
    unsigned short* w1h   = (unsigned short*)carve((size_t)H_N * A_N * HEAD_HID * 2);
    unsigned short* hidbf = (unsigned short*)carve((size_t)B_N * PROJ_HID * 2);
    unsigned short* xbf   = (unsigned short*)carve((size_t)B_N * H_N * 2);
    unsigned short* hbf   = (unsigned short*)carve((size_t)B_N * H_N * 2);
    unsigned short* hpbf  = (unsigned short*)carve((size_t)B_N * H_N * 2);
    (void)ws_size; (void)n_in; (void)in_sizes; (void)out_size;

    // 1) weight + h_prev conversions
    {
        int n = G_N * PROJ_HID;
        k_cvt<<<dim3((n + 255) / 256), dim3(256), 0, stream>>>(proj_w0, w0bf, n);
        n = PROJ_HID * H_N;
        k_cvt<<<dim3((n + 255) / 256), dim3(256), 0, stream>>>(proj_w1, w1bf, n);
        n = 3 * H_N * H_N;
        k_cvt_T<<<dim3((n + 255) / 256), dim3(256), 0, stream>>>(gru_w_ih, wihT, 3 * H_N, H_N);
        k_cvt_T<<<dim3((n + 255) / 256), dim3(256), 0, stream>>>(gru_w_hh, whhT, 3 * H_N, H_N);
        n = A_N * H_N * HEAD_HID;
        k_cvt_headw1<<<dim3((n + 255) / 256), dim3(256), 0, stream>>>(head_w1, w1h);
        n = B_N * H_N;
        k_cvt<<<dim3((n + 255) / 256), dim3(256), 0, stream>>>(h_prev, hpbf, n);
    }

    // 2) obs projection MLP
    k_proj1<<<dim3(B_N / 128), dim3(256), 0, stream>>>(global_obs, w0bf, proj_b0, hidbf);
    k_proj2<<<dim3(B_N / 128, H_N / 64), dim3(256), 0, stream>>>(hidbf, w1bf, proj_b1, xbf);

    // 3) fused GRU step (writes h_next f32 into d_out, bf16 copy for heads)
    k_gru<<<dim3(B_N / 128, H_N / 32), dim3(256), 0, stream>>>(
        xbf, hpbf, h_prev, wihT, whhT, gru_b_ih, gru_b_hh, h_out, hbf);

    // 4) per-agent heads + gather
    k_heads<<<dim3(B_N / 64), dim3(256), 0, stream>>>(
        hbf, w1h, head_b1, head_w2, head_b2, agent_ids, out_sel);
}
